// CASTEBlock_18451179503837
// MI455X (gfx1250) — matrix operand order repaired and checked by reading, not compiled, not run on hardware
//
#include <hip/hip_runtime.h>
#include <stdint.h>

#define BT_N 80
#define B_N 8
#define T_N 10
#define LV 2048
#define CV 256
#define LA 128
#define CA 128
#define PDIM 128
#define KTOP 614
#define LN_EPS_F 1e-5f
#define TILE_L 64

typedef __attribute__((ext_vector_type(16))) __bf16 v16bf;
typedef __attribute__((ext_vector_type(8)))  __bf16 v8bf;
typedef __attribute__((ext_vector_type(8)))  float  v8f;

union FragU { v16bf v; v8bf h[2]; };

// Hardware f32 -> bf16 (RNE) convert; backend emits native cvt ops.
static __device__ __forceinline__ __bf16 f2bf(float x) { return (__bf16)x; }

// A operand: row-major bf16 [M][K], stride in halfs.
// Lane l<16: row=row0+l, K {0..7} then {16..23}; lane>=16: row=row0+l-16, K {8..15} then {24..31}.
static __device__ __forceinline__ v16bf frag_a(const __bf16* base, int stride,
                                               int row0, int kb, int lane) {
  FragU u;
  const __bf16* p = base + (size_t)(row0 + (lane & 15)) * stride + kb + ((lane >> 4) << 3);
  u.h[0] = *(const v8bf*)p;
  u.h[1] = *(const v8bf*)(p + 16);
  return u.v;
}

// B operand: stored row-major as Bt[N][K], stride in halfs.
// Lane l<16: n=n0+l, K {0..7} then {16..23}; lane>=16: n=n0+l-16, K {8..15} then {24..31}.
static __device__ __forceinline__ v16bf frag_b(const __bf16* base, int stride,
                                               int n0, int kb, int lane) {
  FragU u;
  const __bf16* p = base + (size_t)(n0 + (lane & 15)) * stride + kb + ((lane >> 4) << 3);
  u.h[0] = *(const v8bf*)p;
  u.h[1] = *(const v8bf*)(p + 16);
  return u.v;
}

// Async global->LDS 16B copy (CDNA5, ASYNCcnt-tracked).
static __device__ __forceinline__ void cp_async_b128(unsigned lds_off, const void* g) {
  asm volatile("global_load_async_to_lds_b128 %0, %1, off"
               :: "v"(lds_off), "v"(g) : "memory");
}
static __device__ __forceinline__ void cp_async_wait0() {
  asm volatile("s_wait_asynccnt 0x0" ::: "memory");
}

// ---------------------------------------------------------------------------
// K0: weight preconversion. pw_w [d][c] -> bf16 (already B layout for y = dwx @ pw^T).
//     fcv_w [c][e] -> fcvT [e][c] bf16 (B layout for M_bt build, K=c).
__global__ void k0_conv(const float* __restrict__ pw_w, const float* __restrict__ fcv_w,
                        unsigned short* __restrict__ pwbf_u, unsigned short* __restrict__ fcvT_u) {
  __bf16* pwbf = (__bf16*)pwbf_u;
  __bf16* fcvT = (__bf16*)fcvT_u;
  int i = blockIdx.x * blockDim.x + threadIdx.x;  // 65536
  pwbf[i] = f2bf(pw_w[i]);
  int c = i >> 8, e = i & 255;
  fcvT[e * CV + c] = f2bf(fcv_w[c * CV + e]);
}

// ---------------------------------------------------------------------------
// K1: per frame (bt): mean over L, attention logits + softmax, salience,
//     weighted row-sum -> ctx GEMV, AgreementGate MLP -> gate weights.
__global__ void k1_gate_ctx(const float* __restrict__ xv, const float* __restrict__ xa,
                            const float* __restrict__ vp_w, const float* __restrict__ vp_b,
                            const float* __restrict__ ap_w, const float* __restrict__ ap_b,
                            const float* __restrict__ m1_w, const float* __restrict__ m1_b,
                            const float* __restrict__ m2_w, const float* __restrict__ m2_b,
                            const float* __restrict__ fci_w, const float* __restrict__ fci_b,
                            const float* __restrict__ fck_w, const float* __restrict__ fck_b,
                            float* __restrict__ wgate, float* __restrict__ ctx_ws,
                            float* __restrict__ sal_ws) {
  __shared__ float wiL[LV];
  __shared__ float colsum[CV];
  __shared__ float xbar[CV];
  __shared__ float meanV[CV];
  __shared__ float meanA[CA];
  __shared__ float vpr[PDIM];
  __shared__ float apr[PDIM];
  __shared__ float hbuf[PDIM / 2];
  __shared__ float red[256];
  __shared__ float sc[4];

  int bt = blockIdx.x;
  int tid = threadIdx.x;
  int wave = tid >> 5, lane = tid & 31;
  const float* X = xv + (size_t)bt * LV * CV;
  int c0 = lane * 8;

  float fr[8];
  for (int j = 0; j < 8; ++j) fr[j] = fci_w[c0 + j];
  float fcib = fci_b[0];

  colsum[tid] = 0.f;
  xbar[tid] = 0.f;
  __syncthreads();

  // ---- pass 1: column sums, logits, salience ----
  float ca[8];
  for (int j = 0; j < 8; ++j) ca[j] = 0.f;
  for (int l = wave; l < LV; l += 8) {
    const float* row = X + (size_t)l * CV + c0;
    float4 p0 = *(const float4*)row;
    float4 p1 = *(const float4*)(row + 4);
    float v[8] = {p0.x, p0.y, p0.z, p0.w, p1.x, p1.y, p1.z, p1.w};
    float di = 0.f, ds = 0.f;
    for (int j = 0; j < 8; ++j) {
      ca[j] += v[j];
      di += v[j] * fr[j];
      ds += v[j] * v[j];
    }
    for (int m = 16; m; m >>= 1) {
      di += __shfl_xor(di, m, 32);
      ds += __shfl_xor(ds, m, 32);
    }
    if (lane == 0) {
      wiL[l] = di + fcib;
      sal_ws[bt * LV + l] = ds;
    }
  }
  for (int j = 0; j < 8; ++j) atomicAdd(&colsum[c0 + j], ca[j]);
  __syncthreads();
  meanV[tid] = colsum[tid] * (1.f / (float)LV);

  // ---- softmax over wiL (L=2048) ----
  float mx = -3.402823466e38f;
  for (int i = 0; i < 8; ++i) mx = fmaxf(mx, wiL[tid * 8 + i]);
  red[tid] = mx;
  __syncthreads();
  if (tid == 0) {
    float m = red[0];
    for (int i = 1; i < 256; ++i) m = fmaxf(m, red[i]);
    sc[0] = m;
  }
  __syncthreads();
  mx = sc[0];
  float ssum = 0.f;
  for (int i = 0; i < 8; ++i) {
    float e = __expf(wiL[tid * 8 + i] - mx);
    wiL[tid * 8 + i] = e;
    ssum += e;
  }
  red[tid] = ssum;
  __syncthreads();
  if (tid == 0) {
    float s = 0.f;
    for (int i = 0; i < 256; ++i) s += red[i];
    sc[1] = s;
  }
  __syncthreads();
  float inv = 1.f / sc[1];
  for (int i = 0; i < 8; ++i) wiL[tid * 8 + i] *= inv;
  __syncthreads();

  // ---- pass 2 (L2-hot): softmax-weighted row sum ----
  float cb[8];
  for (int j = 0; j < 8; ++j) cb[j] = 0.f;
  for (int l = wave; l < LV; l += 8) {
    float wl = wiL[l];
    const float* row = X + (size_t)l * CV + c0;
    float4 p0 = *(const float4*)row;
    float4 p1 = *(const float4*)(row + 4);
    float v[8] = {p0.x, p0.y, p0.z, p0.w, p1.x, p1.y, p1.z, p1.w};
    for (int j = 0; j < 8; ++j) cb[j] += wl * v[j];
  }
  for (int j = 0; j < 8; ++j) atomicAdd(&xbar[c0 + j], cb[j]);

  // ---- audio mean ----
  if (tid < CA) {
    float s = 0.f;
    const float* A = xa + (size_t)bt * LA * CA + tid;
    for (int r = 0; r < LA; ++r) s += A[r * CA];
    meanA[tid] = s * (1.f / (float)LA);
  }
  __syncthreads();

  // ---- ctx = fck_w @ xbar + fck_b ----
  {
    float acc = fck_b[tid];
    const float* wr = fck_w + (size_t)tid * CV;
    for (int c = 0; c < CV; ++c) acc += wr[c] * xbar[c];
    ctx_ws[bt * CV + tid] = acc;
  }

  // ---- prototypes ----
  if (tid < PDIM) {
    float a1 = vp_b[tid];
    const float* wr = vp_w + (size_t)tid * CV;
    for (int c = 0; c < CV; ++c) a1 += wr[c] * meanV[c];
    vpr[tid] = a1;
    float a2 = ap_b[tid];
    const float* wr2 = ap_w + (size_t)tid * CA;
    for (int c = 0; c < CA; ++c) a2 += wr2[c] * meanA[c];
    apr[tid] = a2;
  }
  __syncthreads();

  if (tid == 0) {
    float dp = 0.f, nv = 0.f, na = 0.f;
    for (int p = 0; p < PDIM; ++p) {
      dp += vpr[p] * apr[p];
      nv += vpr[p] * vpr[p];
      na += apr[p] * apr[p];
    }
    sc[2] = dp / (fmaxf(sqrtf(nv), 1e-12f) * fmaxf(sqrtf(na), 1e-12f));
  }
  __syncthreads();

  // ---- MLP: h = relu(m1 @ feat), feat = [v, a, v-a, v*a] ----
  if (tid < PDIM / 2) {
    float acc = m1_b[tid];
    const float* wr = m1_w + (size_t)tid * 4 * PDIM;
    for (int q = 0; q < PDIM; ++q) {
      float v1 = vpr[q], v2 = apr[q];
      acc += wr[q] * v1 + wr[PDIM + q] * v2 + wr[2 * PDIM + q] * (v1 - v2) +
             wr[3 * PDIM + q] * (v1 * v2);
    }
    hbuf[tid] = fmaxf(acc, 0.f);
  }
  __syncthreads();

  if (tid == 0) {
    float l0 = m2_b[0], l1 = m2_b[1];
    for (int q = 0; q < PDIM / 2; ++q) {
      l0 += m2_w[q] * hbuf[q];
      l1 += m2_w[PDIM / 2 + q] * hbuf[q];
    }
    float ag = sc[2];
    l0 += ag;
    l1 -= ag;
    float m = fmaxf(l0, l1);
    float e0 = __expf(l0 - m), e1 = __expf(l1 - m);
    float s = e0 + e1;
    wgate[bt * 2 + 0] = e0 / s;
    wgate[bt * 2 + 1] = e1 / s;
  }
}

// ---------------------------------------------------------------------------
// K_sel: exact top-k mask per frame by rank counting (ties broken by index).
__global__ void k_sel(const float* __restrict__ sal_ws, float* __restrict__ maskw) {
  __shared__ float s[LV];
  int bt = blockIdx.x, tid = threadIdx.x;
  for (int i = 0; i < 8; ++i) s[tid * 8 + i] = sal_ws[bt * LV + tid * 8 + i];
  __syncthreads();
  for (int i = 0; i < 8; ++i) {
    int l = tid * 8 + i;
    float sl = s[l];
    int cnt = 0;
    for (int j = 0; j < LV; ++j) {
      float sj = s[j];
      cnt += (sj > sl) || (sj == sl && j < l);
    }
    maskw[bt * LV + l] = (cnt < KTOP) ? 1.f : 0.f;
  }
}

// ---------------------------------------------------------------------------
// K2: per frame build M_bt (bf16, stored Bt[d][e] = M[e,d]) via WMMA, plus bias_bt.
//     M[e,d] = sum_c fcv_w[c,e]*ctx[c]*fco_w[d,c] ; GEMM: A=G[d][c]=fco*ctx, B=fcvT[e][c].
__global__ void k2_buildM(const float* __restrict__ fco_w, const float* __restrict__ fcv_b,
                          const float* __restrict__ fco_b, const float* __restrict__ ctx_ws,
                          const unsigned short* __restrict__ fcvT_u,
                          unsigned short* __restrict__ Mt_u, float* __restrict__ biasb) {
  extern __shared__ char smem2[];
  __bf16* G = (__bf16*)smem2;  // 256*256 bf16 = 128KB
  const __bf16* fcvT = (const __bf16*)fcvT_u;
  int bt = blockIdx.x;
  int tid = threadIdx.x;
  const float* ctx = ctx_ws + bt * CV;
  float cc = ctx[tid];
  for (int d = 0; d < CV; ++d) G[d * CV + tid] = f2bf(fco_w[d * CV + tid] * cc);

  float bacc = fco_b[tid];
  for (int c = 0; c < CV; ++c) bacc += fcv_b[c] * ctx[c] * fco_w[tid * CV + c];
  biasb[bt * CV + tid] = bacc;
  __syncthreads();

  int wave = tid >> 5, lane = tid & 31;
  __bf16* MtB = (__bf16*)Mt_u + (size_t)bt * CV * CV;
  for (int s = 0; s < 2; ++s) {
    int mi = (wave * 2 + s) * 16;
    v16bf afr[8];
    for (int e = 0; e < 8; ++e) afr[e] = frag_a(G, CV, mi, e * 32, lane);
    for (int n = 0; n < 16; ++n) {
      v8f acc = {};
      for (int e = 0; e < 8; ++e) {
        v16bf bfr = frag_b(fcvT, CV, n * 16, e * 32, lane);
        acc = __builtin_amdgcn_wmma_f32_16x16x32_bf16(false, afr[e], false, bfr,
                                                      (short)0, acc, false, false);
      }
      int ncol = n * 16 + (lane & 15);
      int rb = (lane >> 4) * 8;
      for (int r = 0; r < 8; ++r)
        MtB[(size_t)(mi + rb + r) * CV + ncol] = f2bf(acc[r]);
    }
  }
}

// ---------------------------------------------------------------------------
// K3: main fused kernel, one (frame, 64-row L tile) per block, 8 waves.
// LDS: Axv 32K | Adw 32K | attn 64K | y 64K | Bst 128K  = 320K (one WG per WGP).
__global__ void __launch_bounds__(256)
k3_main(const float* __restrict__ xv, const unsigned short* __restrict__ Mt_u,
        const float* __restrict__ biasb, const unsigned short* __restrict__ pwbf_u,
        const float* __restrict__ pw_b, const float* __restrict__ dw_w,
        const float* __restrict__ dw_b, const float* __restrict__ spw,
        const float* __restrict__ spb, const float* __restrict__ tmw,
        const float* __restrict__ tmb, const float* __restrict__ wgate,
        const float* __restrict__ maskw, const float* __restrict__ gsp_p,
        const float* __restrict__ gtm_p, const float* __restrict__ gamma_p,
        float* __restrict__ out) {
  extern __shared__ char smem3[];
  __bf16* Axv = (__bf16*)smem3;                           // 64x256 bf16 = 32KB
  __bf16* Adw = Axv + TILE_L * CV;                        // 32KB
  float* attnL = (float*)(smem3 + 2 * TILE_L * CV * 2);   // 64KB
  float* yL = attnL + TILE_L * CV;                        // 64KB
  __bf16* Bst = (__bf16*)(smem3 + 196608);                // 256x256 bf16 = 128KB

  int blk = blockIdx.x;
  int bt = blk >> 5;   // 32 tiles per frame
  int tile = blk & 31;
  int l0 = tile * TILE_L;
  int tid = threadIdx.x;
  int wave = tid >> 5, lane = tid & 31;
  int t = bt % T_N;

  const __bf16* MtB = (const __bf16*)Mt_u + (size_t)bt * CV * CV;
  const __bf16* pwbf = (const __bf16*)pwbf_u;

  // ---- kick off async DMA of M_bt into LDS (overlaps with x staging) ----
  {
    const char* gsrc = (const char*)MtB;
    unsigned lbase = (unsigned)(uintptr_t)Bst;
    for (int i = 0; i < 32; ++i) {
      int idx = i * 256 + tid;  // 8192 x 16B chunks = 128KB
      cp_async_b128(lbase + idx * 16, gsrc + (size_t)idx * 16);
    }
  }

  const float* xc_base = xv + ((size_t)bt * LV + l0) * CV;
  const float* xm_base = xc_base - (size_t)LV * CV;
  const float* xp_base = xc_base + (size_t)LV * CV;
  bool hm = (t > 0), hp = (t < T_N - 1);

  // ---- stage x (bf16) and depthwise-conv tile (bf16) into LDS ----
  for (int i = 0; i < 16; ++i) {
    int e = i * 256 + tid;  // 4096 float4 units
    int row = e >> 6;
    int c4 = (e & 63) * 4;
    size_t off = (size_t)row * CV + c4;
    float4 xc = *(const float4*)(xc_base + off);
    float4 xm = hm ? *(const float4*)(xm_base + off) : make_float4(0.f, 0.f, 0.f, 0.f);
    float4 xp = hp ? *(const float4*)(xp_base + off) : make_float4(0.f, 0.f, 0.f, 0.f);
    const float* pc = (const float*)&xc;
    const float* pm = (const float*)&xm;
    const float* pp = (const float*)&xp;
    for (int j = 0; j < 4; ++j) {
      int c = c4 + j;
      float t0 = dw_w[c * 3 + 0], t1 = dw_w[c * 3 + 1], t2 = dw_w[c * 3 + 2];
      Axv[row * CV + c] = f2bf(pc[j]);
      Adw[row * CV + c] = f2bf(pm[j] * t0 + pc[j] * t1 + pp[j] * t2 + dw_b[c]);
    }
  }
  cp_async_wait0();
  __syncthreads();

  const float* bb = biasb + bt * CV;
  int mi = (wave >> 1) * 16;
  int nhalf = (wave & 1) * 8;

  // GEMM1: attn = x @ M_bt + bias_bt   (A and B both from LDS)
  {
    v16bf afr[8];
    for (int e = 0; e < 8; ++e) afr[e] = frag_a(Axv, CV, mi, e * 32, lane);
    for (int nn = 0; nn < 8; ++nn) {
      int n0 = (nhalf + nn) * 16;
      int ncol = n0 + (lane & 15);
      float bv = bb[ncol];
      v8f acc;
      for (int r = 0; r < 8; ++r) acc[r] = bv;
      for (int e = 0; e < 8; ++e) {
        v16bf bfr = frag_b(Bst, CV, n0, e * 32, lane);
        acc = __builtin_amdgcn_wmma_f32_16x16x32_bf16(false, afr[e], false, bfr,
                                                      (short)0, acc, false, false);
      }
      int rb = (lane >> 4) * 8;
      for (int r = 0; r < 8; ++r) attnL[(mi + rb + r) * CV + ncol] = acc[r];
    }
  }
  // GEMM2: y = dwconv(x) @ pw^T + pw_b   (B from global; shared by all blocks -> L2-hot)
  {
    v16bf afr[8];
    for (int e = 0; e < 8; ++e) afr[e] = frag_a(Adw, CV, mi, e * 32, lane);
    for (int nn = 0; nn < 8; ++nn) {
      int n0 = (nhalf + nn) * 16;
      int ncol = n0 + (lane & 15);
      float bv = pw_b[ncol];
      v8f acc;
      for (int r = 0; r < 8; ++r) acc[r] = bv;
      for (int e = 0; e < 8; ++e) {
        v16bf bfr = frag_b(pwbf, CV, n0, e * 32, lane);
        acc = __builtin_amdgcn_wmma_f32_16x16x32_bf16(false, afr[e], false, bfr,
                                                      (short)0, acc, false, false);
      }
      int rb = (lane >> 4) * 8;
      for (int r = 0; r < 8; ++r) yL[(mi + rb + r) * CV + ncol] = acc[r];
    }
  }
  __syncthreads();

  // epilogue: rowwise LN on both branches, gate-mix with residual, mask, store
  float w0 = wgate[bt * 2 + 0], w1 = wgate[bt * 2 + 1];
  float gsp = gsp_p[0], gtm = gtm_p[0], gma = gamma_p[0];
  int c0 = lane * 8;
  float sw[8], sb[8], tw[8], tb[8];
  for (int j = 0; j < 8; ++j) {
    sw[j] = spw[c0 + j];
    sb[j] = spb[c0 + j];
    tw[j] = tmw[c0 + j];
    tb[j] = tmb[c0 + j];
  }
  for (int q = 0; q < 8; ++q) {
    int r = wave * 8 + q;
    int l = l0 + r;
    float av[8], yv[8];
    float sa = 0.f, sa2 = 0.f, sy = 0.f, sy2 = 0.f;
    for (int j = 0; j < 8; ++j) {
      av[j] = attnL[r * CV + c0 + j];
      sa += av[j];
      sa2 += av[j] * av[j];
      yv[j] = yL[r * CV + c0 + j];
      sy += yv[j];
      sy2 += yv[j] * yv[j];
    }
    for (int m = 16; m; m >>= 1) {
      sa += __shfl_xor(sa, m, 32);
      sa2 += __shfl_xor(sa2, m, 32);
      sy += __shfl_xor(sy, m, 32);
      sy2 += __shfl_xor(sy2, m, 32);
    }
    float ma = sa * (1.f / (float)CV), va = sa2 * (1.f / (float)CV) - ma * ma;
    float my = sy * (1.f / (float)CV), vy = sy2 * (1.f / (float)CV) - my * my;
    float ra = rsqrtf(va + LN_EPS_F), ry = rsqrtf(vy + LN_EPS_F);
    float coef = gma * maskw[bt * LV + l];
    const float* xrow = xc_base + (size_t)r * CV + c0;
    float o[8];
    for (int j = 0; j < 8; ++j) {
      float lna = (av[j] - ma) * ra * sw[j] + sb[j];
      float lny = (yv[j] - my) * ry * tw[j] + tb[j];
      o[j] = coef * (xrow[j] + w0 * gsp * lna + w1 * gtm * lny);
    }
    float* orow = out + ((size_t)(bt * LV + l)) * CV + c0;
    *(float4*)orow = make_float4(o[0], o[1], o[2], o[3]);
    *(float4*)(orow + 4) = make_float4(o[4], o[5], o[6], o[7]);
  }
}

// ---------------------------------------------------------------------------
extern "C" void kernel_launch(void* const* d_in, const int* in_sizes, int n_in,
                              void* d_out, int out_size, void* d_ws, size_t ws_size,
                              hipStream_t stream) {
  const float* xv    = (const float*)d_in[0];
  const float* xa    = (const float*)d_in[1];
  const float* vp_w  = (const float*)d_in[2];
  const float* vp_b  = (const float*)d_in[3];
  const float* ap_w  = (const float*)d_in[4];
  const float* ap_b  = (const float*)d_in[5];
  const float* m1_w  = (const float*)d_in[6];
  const float* m1_b  = (const float*)d_in[7];
  const float* m2_w  = (const float*)d_in[8];
  const float* m2_b  = (const float*)d_in[9];
  const float* fci_w = (const float*)d_in[10];
  const float* fci_b = (const float*)d_in[11];
  const float* fck_w = (const float*)d_in[12];
  const float* fck_b = (const float*)d_in[13];
  const float* fcv_w = (const float*)d_in[14];
  const float* fcv_b = (const float*)d_in[15];
  const float* fco_w = (const float*)d_in[16];
  const float* fco_b = (const float*)d_in[17];
  const float* spw   = (const float*)d_in[18];
  const float* spb   = (const float*)d_in[19];
  const float* gsp   = (const float*)d_in[20];
  const float* dw_w  = (const float*)d_in[21];
  const float* dw_b  = (const float*)d_in[22];
  const float* pw_w  = (const float*)d_in[23];
  const float* pw_b  = (const float*)d_in[24];
  const float* tmw   = (const float*)d_in[25];
  const float* tmb   = (const float*)d_in[26];
  const float* gtm   = (const float*)d_in[27];
  const float* gamma = (const float*)d_in[28];
  float* outp = (float*)d_out;

  char* ws = (char*)d_ws;
  float*          wgate  = (float*)(ws + 0);                  // 640 B
  float*          ctx    = (float*)(ws + 4096);               // 80*256*4
  float*          biasb  = (float*)(ws + 90112);              // 80*256*4
  float*          sal    = (float*)(ws + 176128);             // 80*2048*4
  float*          maskw  = (float*)(ws + 835584);             // 80*2048*4
  unsigned short* pwbf   = (unsigned short*)(ws + 1495040);   // 256*256*2
  unsigned short* fcvT   = (unsigned short*)(ws + 1630208);   // 256*256*2
  unsigned short* Mt     = (unsigned short*)(ws + 1765376);   // 80*256*256*2

  k0_conv<<<dim3(256), dim3(256), 0, stream>>>(pw_w, fcv_w, pwbf, fcvT);

  k1_gate_ctx<<<dim3(BT_N), dim3(256), 0, stream>>>(
      xv, xa, vp_w, vp_b, ap_w, ap_b, m1_w, m1_b, m2_w, m2_b,
      fci_w, fci_b, fck_w, fck_b, wgate, ctx, sal);

  k_sel<<<dim3(BT_N), dim3(256), 0, stream>>>(sal, maskw);

  k2_buildM<<<dim3(BT_N), dim3(256), 131072, stream>>>(
      fco_w, fcv_b, fco_b, ctx, fcvT, Mt, biasb);

  k3_main<<<dim3(BT_N * 32), dim3(256), 327680, stream>>>(
      xv, Mt, biasb, pwbf, pw_b, dw_w, dw_b, spw, spb, tmw, tmb,
      wgate, maskw, gsp, gtm, gamma, outp);
}
